// AdverCE_39994735460869
// MI455X (gfx1250) — matrix operand order repaired and checked by reading, not compiled, not run on hardware
//
#include <hip/hip_runtime.h>
#include <hip/hip_bf16.h>

// ---------------------------------------------------------------------------
// AdverCE on MI455X (gfx1250, wave32):
//   Phase 0: pre-convert q_item_emb to bf16 (once; 12.8MB, L2-resident).
//   Phase 1: fused  S = q_user @ q_item_emb^T + gumbel  (bf16 WMMA, f32 acc).
//            TDM double-buffered 64-item B-panel staging to LDS (8KB/issue),
//            4 sub-tiles (8 WMMAs) per barrier pair, target masking,
//            per-row top-K=128 via LDS min-heaps (one lane per row).
//   Phase 2: merge partial top-K, gather 129 candidates, log-softmax, mean.
// ---------------------------------------------------------------------------

typedef __attribute__((ext_vector_type(16))) __bf16    v16bf;
typedef __attribute__((ext_vector_type(8)))  float     v8f;
typedef __attribute__((ext_vector_type(4)))  unsigned  v4u;
typedef __attribute__((ext_vector_type(8)))  int       v8i;
typedef __attribute__((ext_vector_type(4)))  int       v4i;

#define BB      1024
#define VV      100000
#define DD      64
#define KK      128
#define NCHUNK  16
#define NST     ((VV + 63) / 64)                  // 1563 super-tiles (64 items)
#define SPC     ((NST + NCHUNK - 1) / NCHUNK)     // 98 super-tiles per chunk
#define NEGINF  (-3.4e38f)
#define HSTR    129                               // heap row stride (bank pad)
#define SSTR    65                                // stage row stride (bank pad)

#if __has_builtin(__builtin_amdgcn_tensor_load_to_lds)
#define HAVE_TDM 1
#else
#define HAVE_TDM 0
#endif

// fast -ln(x) for x>0 (bit-trick log2; noise-grade accuracy, no TRANS ops)
__device__ __forceinline__ float fast_nlog(float x) {
  return (float)(1065353216 - __float_as_int(x)) * 8.2629582e-8f;
}

__device__ __forceinline__ float gumbel_fast(unsigned b, unsigned v) {
  unsigned x = (b * 0x9E3779B9u) ^ ((v + 0x7F4A7C15u) * 0x85EBCA6Bu);
  x ^= x >> 16; x *= 0x7FEB352Du;
  x ^= x >> 15; x *= 0x846CA68Bu;
  x ^= x >> 16;
  float u = (float)(x >> 8) * (1.0f / 16777216.0f) + (0.5f / 16777216.0f);
  float e = fast_nlog(u);        // ~Exp(1)
  return fast_nlog(e);           // ~Gumbel(0,1)
}

// ---------------------------------------------------------------------------
// Phase 0: f32 -> bf16 pre-conversion of the item table
// ---------------------------------------------------------------------------
__global__ __launch_bounds__(256) void cvt_bf16(
    const float* __restrict__ x, __bf16* __restrict__ y, int n)
{
  int i = blockIdx.x * 256 + threadIdx.x;
  if (i < n) y[i] = (__bf16)x[i];
}

// ---------------------------------------------------------------------------
// Phase 1: one wave per (16-row block, V-chunk).
// ---------------------------------------------------------------------------
__global__ __launch_bounds__(32) void score_topk_partial(
    const float* __restrict__ q_user, const __bf16* __restrict__ qb,
    const long long* __restrict__ target,
    float* __restrict__ pvals, int* __restrict__ pidx)
{
  __shared__ float hv[16 * HSTR];
  __shared__ int   hi[16 * HSTR];
  __shared__ float stage[16 * SSTR];               // 16 rows x 64 cols (padded)
  __shared__ int   s_tgt[16];
#if HAVE_TDM
  __shared__ __bf16 panel[2][64 * DD];             // 2 x 8KB double buffer
#endif

  const int lane   = threadIdx.x;   // 0..31
  const int bblock = blockIdx.x;    // 0..63
  const int chunk  = blockIdx.y;    // 0..15

  for (int i = lane; i < 16 * HSTR; i += 32) { hv[i] = NEGINF; hi[i] = 0; }
  if (lane < 16) s_tgt[lane] = (int)target[bblock * 16 + lane];
  __syncthreads();

  // ---- A fragments (loop-invariant): bf16 16x32 layout, two K-halves ----
  const int n  = lane & 15;
  const int kh = lane >> 4;
  const float* arow = q_user + (size_t)(bblock * 16 + n) * DD;
  v16bf a0, a1;
#pragma unroll
  for (int e = 0; e < 8; ++e) {
    a0[e]     = (__bf16)arow[ 0 + kh * 16 + e];
    a0[e + 8] = (__bf16)arow[ 8 + kh * 16 + e];
    a1[e]     = (__bf16)arow[32 + kh * 16 + e];
    a1[e + 8] = (__bf16)arow[40 + kh * 16 + e];
  }

  const int s0 = chunk * SPC;
  const int s1 = (s0 + SPC < NST) ? (s0 + SPC) : NST;

#if HAVE_TDM
  // Tensor DMA: 2D tile, 64 rows x 64 bf16 (8KB), row-major, OOB zero-fill.
  const unsigned lds_base = (unsigned)(size_t)(&panel[0][0]);
  v8i g1c;                                        // loop-invariant group 1
  g1c[0] = (int)(1u << 16);                       // data_size = 2 bytes
  g1c[1] = (int)((unsigned)DD << 16);             // tensor_dim0 lo16 = 64
  g1c[2] = (int)(((unsigned)VV & 0xFFFFu) << 16); // dim0 hi16 | dim1 lo16
  g1c[3] = (int)(((unsigned)VV >> 16) | ((unsigned)DD << 16)); // dim1 hi|tile_dim0
  g1c[4] = 64;                                    // tile_dim1 = 64 rows
  g1c[5] = DD;                                    // tensor_dim0_stride lo32
  g1c[6] = 0;
  g1c[7] = 0;
  auto tdm_issue = [&](int st, int buf) {
    unsigned long long ga =
        (unsigned long long)(size_t)(qb + (size_t)st * 64 * DD);
    v4u g0;
    g0[0] = 1u;                                   // count=1, user descriptor
    g0[1] = lds_base + (unsigned)buf * (64 * DD * 2);
    g0[2] = (unsigned)(ga & 0xFFFFFFFFull);
    g0[3] = (unsigned)((ga >> 32) & 0x1FFFFFFull) | (2u << 30);  // type=2
    v4i gz = {0, 0, 0, 0};
#if defined(__clang_major__) && (__clang_major__ >= 23)
    v8i gz8 = {0, 0, 0, 0, 0, 0, 0, 0};
    __builtin_amdgcn_tensor_load_to_lds(g0, g1c, gz, gz, gz8, 0);
#else
    __builtin_amdgcn_tensor_load_to_lds(g0, g1c, gz, gz, 0);
#endif
  };
  tdm_issue(s0, 0);
#endif

  for (int s = s0; s < s1; ++s) {
    const int sbase = s * 64;

#if HAVE_TDM
    const int cur = (s - s0) & 1;
    if (s + 1 < s1) {
      tdm_issue(s + 1, cur ^ 1);
      __builtin_amdgcn_s_wait_tensorcnt(1);  // current panel done, next in flight
    } else {
      __builtin_amdgcn_s_wait_tensorcnt(0);  // drain
    }
#endif
    __syncthreads();

    // ---- 4 sub-tiles, 8 WMMAs, one barrier pair; scheduler interleaves ----
#pragma unroll
    for (int st = 0; st < 4; ++st) {
      const int vbase = sbase + st * 16;
      v16bf b0, b1;
#if HAVE_TDM
      const __bf16* prow = &panel[cur][(st * 16 + n) * DD + kh * 16];
      b0 = *(const v16bf*)(prow);
      b1 = *(const v16bf*)(prow + 32);
#else
      const __bf16* erow = qb + (size_t)(vbase + n) * DD + kh * 16;
      b0 = *(const v16bf*)(erow);
      b1 = *(const v16bf*)(erow + 32);
#endif
      v8f c = {};
      c = __builtin_amdgcn_wmma_f32_16x16x32_bf16(false, a0, false, b0,
                                                  (short)0, c, false, false);
      c = __builtin_amdgcn_wmma_f32_16x16x32_bf16(false, a1, false, b1,
                                                  (short)0, c, false, false);
      // epilogue: gumbel + target/OOB mask, stage scores
#pragma unroll
      for (int j = 0; j < 8; ++j) {
        int mm   = j + 8 * kh;
        int vidx = vbase + n;
        float g  = gumbel_fast((unsigned)(bblock * 16 + mm), (unsigned)vidx);
        float sv = c[j] + g;
        if (vidx == s_tgt[mm] || vidx >= VV) sv = NEGINF;
        stage[mm * SSTR + st * 16 + n] = sv;
      }
    }
    __syncthreads();

    // ---- per-row min-heap top-128 (lane r owns row r; rows in parallel) ----
    if (lane < 16) {
      const int base = lane * HSTR;
      float root = hv[base];
#pragma unroll 1
      for (int nn = 0; nn < 64; ++nn) {
        float sv = stage[lane * SSTR + nn];
        if (sv > root) {
          int p = 0;
          for (;;) {
            int ch = 2 * p + 1;
            if (ch >= KK) break;
            float cv = hv[base + ch];
            if (ch + 1 < KK) {
              float cv2 = hv[base + ch + 1];
              if (cv2 < cv) { cv = cv2; ch = ch + 1; }
            }
            if (cv >= sv) break;
            hv[base + p] = cv; hi[base + p] = hi[base + ch];
            p = ch;
          }
          hv[base + p] = sv; hi[base + p] = sbase + nn;
          root = hv[base];
        }
      }
    }
    __syncthreads();
  }

  // ---- write partial top-K: layout [row_global][chunk][128] ----
  for (int i = lane; i < 16 * KK; i += 32) {
    int r = i >> 7, k = i & (KK - 1);
    size_t off = ((size_t)(bblock * 16 + r) * NCHUNK + chunk) * KK + k;
    pvals[off] = hv[r * HSTR + k];
    pidx [off] = hi[r * HSTR + k];
  }
}

// ---------------------------------------------------------------------------
// Merge 16 partial top-128 lists (2048 candidates) -> global top-128 per row
// ---------------------------------------------------------------------------
__global__ __launch_bounds__(256) void topk_merge(
    const float* __restrict__ pvals, const int* __restrict__ pidx,
    int* __restrict__ out_idx)
{
  __shared__ float cv[2048];
  __shared__ int   ci[2048];
  __shared__ float rv[256];
  __shared__ int   rp[256];
  const int b = blockIdx.x, t = threadIdx.x;

  for (int i = t; i < 2048; i += 256) {
    cv[i] = pvals[(size_t)b * 2048 + i];
    ci[i] = pidx [(size_t)b * 2048 + i];
  }
  __syncthreads();

  for (int k = 0; k < KK; ++k) {
    float bv = NEGINF; int bp = t * 8;
#pragma unroll
    for (int e = 0; e < 8; ++e) {
      float x = cv[t * 8 + e];
      if (x > bv) { bv = x; bp = t * 8 + e; }     // ties: lowest index
    }
    rv[t] = bv; rp[t] = bp;
    __syncthreads();
    for (int off = 128; off > 0; off >>= 1) {
      if (t < off && rv[t + off] > rv[t]) { rv[t] = rv[t + off]; rp[t] = rp[t + off]; }
      __syncthreads();
    }
    if (t == 0) { out_idx[(size_t)b * KK + k] = ci[rp[0]]; cv[rp[0]] = NEGINF; }
    __syncthreads();
  }
}

// ---------------------------------------------------------------------------
// Per-row candidate scores + log-softmax loss (129 candidates)
// ---------------------------------------------------------------------------
__global__ __launch_bounds__(256) void cand_loss(
    const float* __restrict__ p_user, const float* __restrict__ p_item,
    const long long* __restrict__ target, const int* __restrict__ top_idx,
    float* __restrict__ loss_buf)
{
  __shared__ float pu[DD];
  __shared__ float sc[KK + 1];
  __shared__ int   ids[KK + 1];
  const int b = blockIdx.x, t = threadIdx.x;

  if (t < DD) pu[t] = p_user[(size_t)b * DD + t];
  if (t == 0) ids[0] = (int)target[b];
  if (t >= 1 && t <= KK) ids[t] = top_idx[(size_t)b * KK + (t - 1)];
  __syncthreads();

  if (t <= KK) {
    const float* e = p_item + (size_t)ids[t] * DD;
    float acc = 0.f;
#pragma unroll
    for (int d = 0; d < DD; ++d) acc = fmaf(pu[d], e[d], acc);
    sc[t] = acc;
  }
  __syncthreads();

  if (t == 0) {
    float mx = sc[0];
    for (int j = 1; j <= KK; ++j) mx = fmaxf(mx, sc[j]);
    float sum = 0.f;
    for (int j = 0; j <= KK; ++j) sum += __expf(sc[j] - mx);
    loss_buf[b] = -(sc[0] - mx - __logf(sum));
  }
}

// ---------------------------------------------------------------------------
// Deterministic mean over B losses
// ---------------------------------------------------------------------------
__global__ __launch_bounds__(256) void loss_reduce(
    const float* __restrict__ loss_buf, float* __restrict__ out)
{
  __shared__ float r[256];
  const int t = threadIdx.x;
  float acc = 0.f;
  for (int i = t; i < BB; i += 256) acc += loss_buf[i];
  r[t] = acc;
  __syncthreads();
  for (int off = 128; off > 0; off >>= 1) {
    if (t < off) r[t] += r[t + off];
    __syncthreads();
  }
  if (t == 0) out[0] = r[0] * (1.0f / (float)BB);
}

// ---------------------------------------------------------------------------
extern "C" void kernel_launch(void* const* d_in, const int* in_sizes, int n_in,
                              void* d_out, int out_size, void* d_ws, size_t ws_size,
                              hipStream_t stream)
{
  (void)in_sizes; (void)n_in; (void)out_size; (void)ws_size;

  const float*     q_user = (const float*)d_in[0];
  const float*     q_item = (const float*)d_in[1];
  const float*     p_user = (const float*)d_in[2];
  const float*     p_item = (const float*)d_in[3];
  const long long* target = (const long long*)d_in[4];

  char* ws = (char*)d_ws;
  __bf16* qb   = (__bf16*)ws;                               // V*D bf16 (12.8MB)
  size_t  off  = (size_t)VV * DD * 2;
  float*  pvals = (float*)(ws + off);                       // B*16*128 f32
  off += (size_t)BB * NCHUNK * KK * 4;
  int*    pidx  = (int*)(ws + off);                         // B*16*128 i32
  off += (size_t)BB * NCHUNK * KK * 4;
  int*    tidx  = (int*)(ws + off);                         // B*128 i32
  off += (size_t)BB * KK * 4;
  float*  lossb = (float*)(ws + off);                       // B f32

  cvt_bf16<<<(VV * DD + 255) / 256, 256, 0, stream>>>(q_item, qb, VV * DD);

  dim3 g1(BB / 16, NCHUNK);
  score_topk_partial<<<g1, 32, 0, stream>>>(q_user, qb, target, pvals, pidx);
  topk_merge<<<BB, 256, 0, stream>>>(pvals, pidx, tidx);
  cand_loss<<<BB, 256, 0, stream>>>(p_user, p_item, target, tidx, lossb);
  loss_reduce<<<1, 256, 0, stream>>>(lossb, (float*)d_out);
}
